// SSMBottleneck_71725953843485
// MI455X (gfx1250) — matrix operand order repaired and checked by reading, not compiled, not run on hardware
//
#include <hip/hip_runtime.h>
#include <cstdint>

#define BATCH   4
#define SEQLEN  4096
#define NTOK    (BATCH*SEQLEN)
#define DMODEL  384
#define DINNER  768
#define DSTATE  64
#define NH      12
#define HD      64
#define NCH     64
#define CDIM    896
#define DPROJ   1676
#define DTOFF   1664

typedef __bf16 bf16;
typedef __attribute__((ext_vector_type(16))) __bf16 v16bf;
typedef __attribute__((ext_vector_type(8)))  __bf16 v8bf;
typedef __attribute__((ext_vector_type(8)))  float  v8f;

// direction mapping: token index t (scan order) -> natural row-major position
template<int D>
__device__ __forceinline__ int dmapT(int t) {
  if constexpr (D == 1)      return SEQLEN - 1 - t;
  else if constexpr (D == 2) return ((t & 63) << 6) | (t >> 6);
  else if constexpr (D == 3) { int u = SEQLEN - 1 - t; return ((u & 63) << 6) | (u >> 6); }
  else return t;
}

// ---- WMMA fragment loaders (contiguous k order: k = 16*(lane/16) + elem, wave32) ----
// A (16x32): row-major s[row][k]; per lane one contiguous 16-elem (32B) run.
__device__ __forceinline__ v16bf fragA(const bf16* s, int ld, int row, int k0) {
  int lane = threadIdx.x & 31;
  int r  = row + (lane & 15);
  int hs = (lane >> 4) << 4;                       // 0 or 16
  const bf16* p = s + r * ld + k0 + hs;
  v8bf lo = *(const v8bf*)p;
  v8bf hi = *(const v8bf*)(p + 8);
  return __builtin_shufflevector(lo, hi, 0,1,2,3,4,5,6,7,8,9,10,11,12,13,14,15);
}
// B (32x16, KxN) from [col][k]-stored operand: one contiguous 16-elem (32B) run.
__device__ __forceinline__ v16bf fragBT(const bf16* s, int ld, int k0, int col) {
  int lane = threadIdx.x & 31;
  int c  = col + (lane & 15);
  int hs = (lane >> 4) << 4;                       // 0 or 16
  const bf16* p = s + c * ld + k0 + hs;
  v8bf lo = *(const v8bf*)p;
  v8bf hi = *(const v8bf*)(p + 8);
  return __builtin_shufflevector(lo, hi, 0,1,2,3,4,5,6,7,8,9,10,11,12,13,14,15);
}
// B (32x16, KxN) from row-major s[k][col]: strided (unavoidable).
__device__ __forceinline__ v16bf fragB_strided(const bf16* s, int ld, int k0, int col) {
  int lane = threadIdx.x & 31;
  int c  = col + (lane & 15);
  int hs = (lane >> 4) << 4;
  v16bf b;
#pragma unroll
  for (int v = 0; v < 8; ++v) {
    int k = k0 + 2*v + hs;
    b[2*v]   = s[k*ld + c];
    b[2*v+1] = s[(k+1)*ld + c];
  }
  return b;
}
// A = sᵀ with per-k scale: A[row][k] = s[k*ld + row] * scale[k]  (strided)
__device__ __forceinline__ v16bf fragAT_scaled(const bf16* s, int ld, int row, int k0,
                                               const float* scale) {
  int lane = threadIdx.x & 31;
  int r  = row + (lane & 15);
  int hs = (lane >> 4) << 4;
  v16bf a;
#pragma unroll
  for (int v = 0; v < 8; ++v) {
    int k = k0 + 2*v + hs;
    a[2*v]   = (bf16)((float)s[k*ld + r]     * scale[k]);
    a[2*v+1] = (bf16)((float)s[(k+1)*ld + r] * scale[k+1]);
  }
  return a;
}

#define WMMA_BF16(A, B, C) \
  __builtin_amdgcn_wmma_f32_16x16x32_bf16(false, (A), false, (B), (short)0, (C), false, false)

// ---------------- elementwise / prep kernels ----------------
// convert + transpose: in[K][N] f32 -> out[N][K] bf16
__global__ void f2bf_t_kernel(const float* __restrict__ in, bf16* __restrict__ out,
                              int K, int N) {
  int i = blockIdx.x * blockDim.x + threadIdx.x;
  if (i >= K * N) return;
  int n = i % N, k = i / N;
  out[(size_t)n * K + k] = (bf16)in[i];
}

__global__ void nchw_to_nlc(const float* __restrict__ x, float* __restrict__ res) {
  int idx = blockIdx.x * blockDim.x + threadIdx.x;
  if (idx >= NTOK * DMODEL) return;
  int c  = idx % DMODEL;
  int l  = (idx / DMODEL) % SEQLEN;
  int bb = idx / (DMODEL * SEQLEN);
  res[idx] = x[((size_t)bb * DMODEL + c) * SEQLEN + l];
}

__global__ void nlc_to_nchw(const float* __restrict__ res, float* __restrict__ out) {
  int idx = blockIdx.x * blockDim.x + threadIdx.x;
  if (idx >= NTOK * DMODEL) return;
  int l  = idx % SEQLEN;
  int c  = (idx / SEQLEN) % DMODEL;
  int bb = idx / (SEQLEN * DMODEL);
  out[idx] = res[((size_t)bb * SEQLEN + l) * DMODEL + c];
}

__global__ __launch_bounds__(128)
void layernorm_kernel(const float* __restrict__ res, const float* __restrict__ w,
                      const float* __restrict__ b, bf16* __restrict__ nx) {
  int row = blockIdx.x;
  const float* xr = res + (size_t)row * DMODEL;
  __shared__ float red[128];
  float v[3]; float s = 0.f, s2 = 0.f;
#pragma unroll
  for (int i = 0; i < 3; ++i) {
    int c = threadIdx.x + i * 128;
    v[i] = xr[c];
    s += v[i]; s2 += v[i] * v[i];
  }
  red[threadIdx.x] = s; __syncthreads();
  for (int off = 64; off; off >>= 1) { if (threadIdx.x < off) red[threadIdx.x] += red[threadIdx.x + off]; __syncthreads(); }
  float mean = red[0] / DMODEL; __syncthreads();
  red[threadIdx.x] = s2; __syncthreads();
  for (int off = 64; off; off >>= 1) { if (threadIdx.x < off) red[threadIdx.x] += red[threadIdx.x + off]; __syncthreads(); }
  float var = red[0] / DMODEL - mean * mean;
  float inv = rsqrtf(var + 1e-5f);
#pragma unroll
  for (int i = 0; i < 3; ++i) {
    int c = threadIdx.x + i * 128;
    nx[(size_t)row * DMODEL + c] = (bf16)((v[i] - mean) * inv * w[c] + b[c]);
  }
}

// ---------------- bf16 WMMA GEMM: Out[M=NTOK x N] (+)= gather(A) @ Wt' ----------------
// Wt stored transposed [N][K]. Block tile 128M x 64N, software-pipelined staging.
template<int K, int DIRIN, int DIROUT, bool ACCUM>
__global__ __launch_bounds__(256)
void gemm_wmma(const bf16* __restrict__ A, const bf16* __restrict__ Wt,
               int N, float* __restrict__ Out) {
  __shared__ __align__(16) bf16 Ash[128 * 40];
  __shared__ __align__(16) bf16 Wsh[64 * 40];
  int n0 = blockIdx.x * 64;
  int m0 = blockIdx.y * 128;
  int tid = threadIdx.x;
  int w  = tid >> 5;
  int wm = w & 3;    // two 16-row strips: 16*wm and 64+16*wm
  int wn = w >> 2;   // 32-col strip
  v8f acc00 = {}, acc01 = {}, acc10 = {}, acc11 = {};

  // staging assignments (hoisted out of K loop)
  int rA  = tid >> 1;              // 0..127
  int cgA = (tid & 1) * 16;        // 0,16
  int rowA = m0 + rA;
  int srcA = ((rowA >> 12) << 12) + dmapT<DIRIN>(rowA & (SEQLEN - 1));
  const bf16* gA = A + (size_t)srcA * K + cgA;
  bf16* sA = &Ash[rA * 40 + cgA];

  int rW  = tid >> 2;              // 0..63
  int cgW = (tid & 3) * 8;
  int gn = n0 + rW;
  const bf16* gW = Wt + (size_t)gn * K + cgW;
  bool wok = (gn < N);
  bf16* sW = &Wsh[rW * 40 + cgW];

  // prologue prefetch
  uint4 av0 = *(const uint4*)(gA);
  uint4 av1 = *(const uint4*)(gA + 8);
  uint4 wv = {0, 0, 0, 0};
  if (wok) wv = *(const uint4*)(gW);

  for (int ko = 0; ko < K; ko += 32) {
    *(uint4*)sA       = av0;
    *(uint4*)(sA + 8) = av1;
    *(uint4*)sW       = wv;
    __syncthreads();
    if (ko + 32 < K) {   // prefetch next K-slab while WMMAs run
      av0 = *(const uint4*)(gA + ko + 32);
      av1 = *(const uint4*)(gA + ko + 40);
      if (wok) wv = *(const uint4*)(gW + ko + 32);
    }
    v16bf a0 = fragA(Ash, 40, wm * 16, 0);
    v16bf a1 = fragA(Ash, 40, 64 + wm * 16, 0);
    v16bf b0 = fragBT(Wsh, 40, 0, wn * 32);
    v16bf b1 = fragBT(Wsh, 40, 0, wn * 32 + 16);
    acc00 = WMMA_BF16(a0, b0, acc00);
    acc01 = WMMA_BF16(a0, b1, acc01);
    acc10 = WMMA_BF16(a1, b0, acc10);
    acc11 = WMMA_BF16(a1, b1, acc11);
    __syncthreads();
  }

  int lane = tid & 31;
  int hs = lane >> 4;
  int cbase = n0 + wn * 32;
  v8f accs[2][2] = { { acc00, acc01 }, { acc10, acc11 } };
#pragma unroll
  for (int mi = 0; mi < 2; ++mi)
#pragma unroll
    for (int ni = 0; ni < 2; ++ni)
#pragma unroll
      for (int r = 0; r < 8; ++r) {
        int m  = m0 + mi * 64 + wm * 16 + r + 8 * hs;
        int orow = ((m >> 12) << 12) + dmapT<DIROUT>(m & (SEQLEN - 1));
        int cc = cbase + ni * 16 + (lane & 15);
        if (cc < N) {
          float* o = Out + (size_t)orow * N + cc;
          if (ACCUM) *o += accs[mi][ni][r]; else *o = accs[mi][ni][r];
        }
      }
}

// ---------------- dt: softplus, a = A*dt, per-chunk inclusive cumsum ----------------
__global__ __launch_bounds__(64)
void dt_kernel(const float* __restrict__ zx, const float* __restrict__ dtbias,
               const float* __restrict__ Alog,
               float* __restrict__ dtsp, float* __restrict__ acsb, float* __restrict__ chsum) {
  int bc = blockIdx.x;
  int c  = bc & (NCH - 1);
  int bb = bc >> 6;
  int t  = threadIdx.x;
  int row = bb * SEQLEN + c * 64 + t;
  __shared__ float sh[64];
  for (int h = 0; h < NH; ++h) {
    float x = zx[(size_t)row * DPROJ + DTOFF + h] + dtbias[h];
    float sp = (x > 20.f) ? x : log1pf(__expf(x));
    float a  = -__expf(Alog[h]) * sp;
    sh[t] = a; __syncthreads();
    for (int off = 1; off < 64; off <<= 1) {
      float v = sh[t];
      if (t >= off) v += sh[t - off];
      __syncthreads();
      sh[t] = v;
      __syncthreads();
    }
    float cs = sh[t];
    dtsp[(size_t)row * NH + h] = sp;
    acsb[(size_t)row * NH + h] = cs;
    if (t == 63) chsum[(bb * NH + h) * NCH + c] = cs;
    __syncthreads();
  }
}

// ---------------- depthwise causal conv(4) + SiLU ----------------
__global__ void conv_silu_kernel(const float* __restrict__ zx, const float* __restrict__ cw,
                                 const float* __restrict__ cb,
                                 bf16* __restrict__ xb, bf16* __restrict__ Bb, bf16* __restrict__ Cb) {
  int idx = blockIdx.x * blockDim.x + threadIdx.x;
  if (idx >= NTOK * CDIM) return;
  int ch = idx % CDIM;
  int t  = (idx / CDIM) % SEQLEN;
  int bb = idx / (CDIM * SEQLEN);
  const float* w = cw + ch * 4;
  float acc = cb[ch];
  size_t base = ((size_t)bb * SEQLEN) * DPROJ + DINNER + ch;
#pragma unroll
  for (int k = 0; k < 4; ++k) {
    int ti = t - 3 + k;
    if (ti >= 0) acc += w[k] * zx[base + (size_t)ti * DPROJ];
  }
  float s = acc / (1.f + __expf(-acc));
  size_t row = (size_t)bb * SEQLEN + t;
  if (ch < DINNER)            xb[row * DINNER + ch]        = (bf16)s;
  else if (ch < DINNER + 64)  Bb[row * 64 + (ch - DINNER)] = (bf16)s;
  else                        Cb[row * 64 + (ch - DINNER - 64)] = (bf16)s;
}

// ---------------- SSD per (b,chunk,head) ----------------
__global__ __launch_bounds__(128)
void ssd_diag_states(const bf16* __restrict__ Xb, const bf16* __restrict__ Bb,
                     const bf16* __restrict__ Cb, const float* __restrict__ dtsp,
                     const float* __restrict__ acs,
                     float* __restrict__ Ybuf, float* __restrict__ Sbuf) {
  int bx = blockIdx.x;
  int h  = bx % NH;
  int c  = (bx / NH) % NCH;
  int bb = bx / (NH * NCH);
  int tid = threadIdx.x;
  int w = tid >> 5;
  __shared__ __align__(16) bf16 Csh[64 * 72], Bsh[64 * 72], Xsh[64 * 72], Ssh[64 * 72];
  __shared__ float acs_sh[64], dec_sh[64], dts_sh[64];

  if (tid < 64) {
    size_t gi = ((size_t)(bb * SEQLEN + c * 64 + tid)) * NH + h;
    acs_sh[tid] = acs[gi];
    dts_sh[tid] = dtsp[gi];
  }
  __syncthreads();
  if (tid < 64) dec_sh[tid] = __expf(acs_sh[63] - acs_sh[tid]);
  { // stage tiles (vectorized)
    int r  = tid >> 1;
    int h0 = (tid & 1) * 32;
    size_t grow = (size_t)bb * SEQLEN + c * 64 + r;
    const bf16* xp = Xb + (grow * NH + h) * HD + h0;
    const bf16* bp = Bb + grow * 64 + h0;
    const bf16* cp = Cb + grow * 64 + h0;
    float ds = dts_sh[r];
#pragma unroll
    for (int q = 0; q < 2; ++q) {
      *(uint4*)&Bsh[r * 72 + h0 + q * 16] = *(const uint4*)(bp + q * 16);
      *(uint4*)&Csh[r * 72 + h0 + q * 16] = *(const uint4*)(cp + q * 16);
    }
#pragma unroll
    for (int q = 0; q < 4; ++q) {
      v8bf xv = *(const v8bf*)(xp + q * 8);
      v8bf o;
#pragma unroll
      for (int i = 0; i < 8; ++i) o[i] = (bf16)((float)xv[i] * ds);
      *(v8bf*)&Xsh[r * 72 + h0 + q * 8] = o;
    }
  }
  __syncthreads();

  v8f zero = {};
  // S = C (l x n) @ Bᵀ (n x s)  — both operands contiguous
  v8f sA[4] = { zero, zero, zero, zero };
#pragma unroll
  for (int j = 0; j < 4; ++j)
#pragma unroll
    for (int ks = 0; ks < 64; ks += 32) {
      v16bf a  = fragA(Csh, 72, w * 16, ks);
      v16bf bt = fragBT(Bsh, 72, ks, j * 16);
      sA[j] = WMMA_BF16(a, bt, sA[j]);
    }
  int lane = tid & 31, hs = lane >> 4;
  // branchless triangular mask + decay (exp computed unconditionally, cndmask select)
#pragma unroll
  for (int j = 0; j < 4; ++j)
#pragma unroll
    for (int r = 0; r < 8; ++r) {
      int m = w * 16 + r + 8 * hs;
      int n = j * 16 + (lane & 15);
      float e = __expf(acs_sh[m] - acs_sh[n]);
      e = (n <= m) ? e : 0.f;
      Ssh[m * 72 + n] = (bf16)(sA[j][r] * e);
    }
  __syncthreads();

  // Y_diag = S (l x s) @ X̂ (s x p)
  v8f yA[4] = { zero, zero, zero, zero };
#pragma unroll
  for (int j = 0; j < 4; ++j)
#pragma unroll
    for (int ks = 0; ks < 64; ks += 32) {
      v16bf a = fragA(Ssh, 72, w * 16, ks);
      v16bf b = fragB_strided(Xsh, 72, ks, j * 16);
      yA[j] = WMMA_BF16(a, b, yA[j]);
    }
#pragma unroll
  for (int j = 0; j < 4; ++j)
#pragma unroll
    for (int r = 0; r < 8; ++r) {
      int m = w * 16 + r + 8 * hs;
      int p = j * 16 + (lane & 15);
      size_t yi = (((size_t)(bb * SEQLEN + c * 64 + m)) * NH + h) * HD + p;
      Ybuf[yi] = yA[j][r];
    }

  // states = (X̂ * decay)ᵀ (p x l) @ B (l x n)
  v8f stA[4] = { zero, zero, zero, zero };
#pragma unroll
  for (int j = 0; j < 4; ++j)
#pragma unroll
    for (int ks = 0; ks < 64; ks += 32) {
      v16bf a = fragAT_scaled(Xsh, 72, w * 16, ks, dec_sh);
      v16bf b = fragB_strided(Bsh, 72, ks, j * 16);
      stA[j] = WMMA_BF16(a, b, stA[j]);
    }
  size_t sbase = ((size_t)((bb * NCH + c) * NH + h)) << 12;
#pragma unroll
  for (int j = 0; j < 4; ++j)
#pragma unroll
    for (int r = 0; r < 8; ++r) {
      int p = w * 16 + r + 8 * hs;
      int n = j * 16 + (lane & 15);
      Sbuf[sbase + p * 64 + n] = stA[j][r];
    }
}

// ---------------- inter-chunk recurrence ----------------
__global__ __launch_bounds__(256)
void chain_kernel(const float* __restrict__ Sbuf, const float* __restrict__ chsum,
                  float* __restrict__ Pbuf) {
  int bh = blockIdx.x;
  int h  = bh % NH;
  int bb = bh / NH;
  float pr[16];
#pragma unroll
  for (int j = 0; j < 16; ++j) pr[j] = 0.f;
  for (int c = 0; c < NCH; ++c) {
    size_t base = ((size_t)((bb * NCH + c) * NH + h)) << 12;
    float dc = __expf(chsum[(bb * NH + h) * NCH + c]);
#pragma unroll
    for (int j = 0; j < 16; ++j) {
      int e = threadIdx.x + j * 256;
      Pbuf[base + e] = pr[j];
      pr[j] = (pr[j] + Sbuf[base + e]) * dc;
    }
  }
}

// ---------------- Y_off = (C * exp(Acs)) @ prevᵀ ; y += Y_off + D*x ----------------
__global__ __launch_bounds__(128)
void ssd_yoff(const bf16* __restrict__ Xb, const bf16* __restrict__ Cb,
              const float* __restrict__ acs, const float* __restrict__ Pbuf,
              const float* __restrict__ Dp, float* __restrict__ Ybuf) {
  int bx = blockIdx.x;
  int h  = bx % NH;
  int c  = (bx / NH) % NCH;
  int bb = bx / (NH * NCH);
  int tid = threadIdx.x;
  int w = tid >> 5;
  __shared__ __align__(16) bf16 Cesh[64 * 72], Psh[64 * 72];
  __shared__ float acs_sh[64];

  if (tid < 64) acs_sh[tid] = acs[((size_t)(bb * SEQLEN + c * 64 + tid)) * NH + h];
  __syncthreads();
  {
    int r  = tid >> 1;
    int h0 = (tid & 1) * 32;
    size_t grow = (size_t)bb * SEQLEN + c * 64 + r;
    const bf16* cp = Cb + grow * 64 + h0;
    size_t pbase = (((size_t)((bb * NCH + c) * NH + h)) << 12) + (size_t)r * 64 + h0;
    float e = __expf(acs_sh[r]);
#pragma unroll
    for (int q = 0; q < 4; ++q) {
      v8bf cv = *(const v8bf*)(cp + q * 8);
      v8bf o;
#pragma unroll
      for (int i = 0; i < 8; ++i) o[i] = (bf16)((float)cv[i] * e);
      *(v8bf*)&Cesh[r * 72 + h0 + q * 8] = o;
      float4 f0 = *(const float4*)&Pbuf[pbase + q * 8];
      float4 f1 = *(const float4*)&Pbuf[pbase + q * 8 + 4];
      v8bf pv;
      pv[0] = (bf16)f0.x; pv[1] = (bf16)f0.y; pv[2] = (bf16)f0.z; pv[3] = (bf16)f0.w;
      pv[4] = (bf16)f1.x; pv[5] = (bf16)f1.y; pv[6] = (bf16)f1.z; pv[7] = (bf16)f1.w;
      *(v8bf*)&Psh[r * 72 + h0 + q * 8] = pv;
    }
  }
  __syncthreads();

  v8f zero = {};
  v8f yA[4] = { zero, zero, zero, zero };
#pragma unroll
  for (int j = 0; j < 4; ++j)
#pragma unroll
    for (int ks = 0; ks < 64; ks += 32) {
      v16bf a  = fragA(Cesh, 72, w * 16, ks);
      v16bf bt = fragBT(Psh, 72, ks, j * 16);   // B[k=n][col=p] = Psh[p][n]
      yA[j] = WMMA_BF16(a, bt, yA[j]);
    }
  int lane = tid & 31, hs = lane >> 4;
  float Dv = Dp[h];
#pragma unroll
  for (int j = 0; j < 4; ++j)
#pragma unroll
    for (int r = 0; r < 8; ++r) {
      int m = w * 16 + r + 8 * hs;
      int p = j * 16 + (lane & 15);
      size_t grow = (size_t)bb * SEQLEN + c * 64 + m;
      size_t yi = (grow * NH + h) * HD + p;
      float xv = (float)Xb[(grow * NH + h) * HD + p];
      Ybuf[yi] += yA[j][r] + Dv * xv;
    }
}

// ---------------- gate (y * silu(z)) + RMSNorm -> bf16 ----------------
__global__ __launch_bounds__(256)
void gate_norm_kernel(const float* __restrict__ Ybuf, const float* __restrict__ zx,
                      const float* __restrict__ normw, bf16* __restrict__ outbf) {
  int row = blockIdx.x;
  const float* y = Ybuf + (size_t)row * DINNER;
  const float* z = zx + (size_t)row * DPROJ;
  __shared__ float red[256];
  float g[3]; float ss = 0.f;
#pragma unroll
  for (int i = 0; i < 3; ++i) {
    int c = threadIdx.x + i * 256;
    float zz = z[c];
    float gv = y[c] * (zz / (1.f + __expf(-zz)));
    g[i] = gv; ss += gv * gv;
  }
  red[threadIdx.x] = ss; __syncthreads();
  for (int off = 128; off; off >>= 1) { if (threadIdx.x < off) red[threadIdx.x] += red[threadIdx.x + off]; __syncthreads(); }
  float inv = rsqrtf(red[0] / DINNER + 1e-5f);
#pragma unroll
  for (int i = 0; i < 3; ++i) {
    int c = threadIdx.x + i * 256;
    outbf[(size_t)row * DINNER + c] = (bf16)(g[i] * inv * normw[c]);
  }
}

// =====================================================================
extern "C" void kernel_launch(void* const* d_in, const int* in_sizes, int n_in,
                              void* d_out, int out_size, void* d_ws, size_t ws_size,
                              hipStream_t stream) {
  const float* x = (const float*)d_in[0];

  size_t off = 0;
  auto alloc = [&](size_t bytes) -> void* {
    void* p = (char*)d_ws + off;
    off += (bytes + 255) & ~(size_t)255;
    return p;
  };
  float* res    = (float*)alloc((size_t)NTOK * DMODEL * 4);
  bf16*  nx     = (bf16*) alloc((size_t)NTOK * DMODEL * 2);
  bf16*  win[2], *wout[2];
  for (int l = 0; l < 2; ++l) {
    win[l]  = (bf16*)alloc((size_t)DMODEL * DPROJ * 2);   // stored [DPROJ][DMODEL]
    wout[l] = (bf16*)alloc((size_t)DINNER * DMODEL * 2);  // stored [DMODEL][DINNER]
  }
  float* zx     = (float*)alloc((size_t)NTOK * DPROJ * 4);
  bf16*  xb     = (bf16*) alloc((size_t)NTOK * DINNER * 2);
  bf16*  Bb     = (bf16*) alloc((size_t)NTOK * 64 * 2);
  bf16*  Cb     = (bf16*) alloc((size_t)NTOK * 64 * 2);
  float* dtsp   = (float*)alloc((size_t)NTOK * NH * 4);
  float* acs    = (float*)alloc((size_t)NTOK * NH * 4);
  float* chsum  = (float*)alloc((size_t)BATCH * NH * NCH * 4);
  float* Sbuf   = (float*)alloc((size_t)BATCH * NCH * NH * HD * DSTATE * 4);
  float* Pbuf   = (float*)alloc((size_t)BATCH * NCH * NH * HD * DSTATE * 4);
  float* Ybuf   = (float*)alloc((size_t)NTOK * DINNER * 4);
  bf16*  normed = (bf16*)Sbuf;  // overlay: Sbuf dead before gate_norm runs

  for (int l = 0; l < 2; ++l) {
    const float* ip = (const float*)d_in[1 + l * 10 + 2];
    const float* op = (const float*)d_in[1 + l * 10 + 9];
    int n1 = DMODEL * DPROJ, n2 = DINNER * DMODEL;
    f2bf_t_kernel<<<(n1 + 255) / 256, 256, 0, stream>>>(ip, win[l], DMODEL, DPROJ);
    f2bf_t_kernel<<<(n2 + 255) / 256, 256, 0, stream>>>(op, wout[l], DINNER, DMODEL);
  }
  {
    int n = NTOK * DMODEL;
    nchw_to_nlc<<<(n + 255) / 256, 256, 0, stream>>>(x, res);
  }

  dim3 gIn((DPROJ + 63) / 64, NTOK / 128);
  dim3 gOut(DMODEL / 64, NTOK / 128);

  for (int layer = 0; layer < 2; ++layer) {
    int lb = 1 + layer * 10;
    const float* ln_w    = (const float*)d_in[lb + 0];
    const float* ln_b    = (const float*)d_in[lb + 1];
    const float* conv_w  = (const float*)d_in[lb + 3];
    const float* conv_b  = (const float*)d_in[lb + 4];
    const float* dt_bias = (const float*)d_in[lb + 5];
    const float* A_log   = (const float*)d_in[lb + 6];
    const float* Dparam  = (const float*)d_in[lb + 7];
    const float* norm_w  = (const float*)d_in[lb + 8];

    layernorm_kernel<<<NTOK, 128, 0, stream>>>(res, ln_w, ln_b, nx);

    for (int dir = 0; dir < 4; ++dir) {
      switch (dir) {  // in_proj GEMM (gathered input rows)
        case 0: gemm_wmma<DMODEL, 0, -1, false><<<gIn, 256, 0, stream>>>(nx, win[layer], DPROJ, zx); break;
        case 1: gemm_wmma<DMODEL, 1, -1, false><<<gIn, 256, 0, stream>>>(nx, win[layer], DPROJ, zx); break;
        case 2: gemm_wmma<DMODEL, 2, -1, false><<<gIn, 256, 0, stream>>>(nx, win[layer], DPROJ, zx); break;
        case 3: gemm_wmma<DMODEL, 3, -1, false><<<gIn, 256, 0, stream>>>(nx, win[layer], DPROJ, zx); break;
      }
      dt_kernel<<<BATCH * NCH, 64, 0, stream>>>(zx, dt_bias, A_log, dtsp, acs, chsum);
      {
        int n = NTOK * CDIM;
        conv_silu_kernel<<<(n + 255) / 256, 256, 0, stream>>>(zx, conv_w, conv_b, xb, Bb, Cb);
      }
      ssd_diag_states<<<BATCH * NCH * NH, 128, 0, stream>>>(xb, Bb, Cb, dtsp, acs, Ybuf, Sbuf);
      chain_kernel<<<BATCH * NH, 256, 0, stream>>>(Sbuf, chsum, Pbuf);
      ssd_yoff<<<BATCH * NCH * NH, 128, 0, stream>>>(xb, Cb, acs, Pbuf, Dparam, Ybuf);
      gate_norm_kernel<<<NTOK, 256, 0, stream>>>(Ybuf, zx, norm_w, normed);
      switch (dir) {  // out_proj GEMM (scatter-accumulate into residual)
        case 0: gemm_wmma<DINNER, -1, 0, true><<<gOut, 256, 0, stream>>>(normed, wout[layer], DMODEL, res); break;
        case 1: gemm_wmma<DINNER, -1, 1, true><<<gOut, 256, 0, stream>>>(normed, wout[layer], DMODEL, res); break;
        case 2: gemm_wmma<DINNER, -1, 2, true><<<gOut, 256, 0, stream>>>(normed, wout[layer], DMODEL, res); break;
        case 3: gemm_wmma<DINNER, -1, 3, true><<<gOut, 256, 0, stream>>>(normed, wout[layer], DMODEL, res); break;
      }
    }
  }
  {
    int n = NTOK * DMODEL;
    nlc_to_nchw<<<(n + 255) / 256, 256, 0, stream>>>(res, (float*)d_out);
  }
  (void)in_sizes; (void)n_in; (void)out_size; (void)ws_size;
}
